// Transformer_67963562492139
// MI455X (gfx1250) — matrix operand order repaired and checked by reading, not compiled, not run on hardware
//
#include <hip/hip_runtime.h>
#include <math.h>

typedef unsigned short u16;
typedef __attribute__((ext_vector_type(16))) __bf16 v16bf;
typedef __attribute__((ext_vector_type(8)))  float  v8f;
typedef __attribute__((ext_vector_type(4)))  unsigned int v4u;
typedef __attribute__((ext_vector_type(8)))  int v8i;
typedef __attribute__((ext_vector_type(4)))  int v4i;

#define BM 128
#define BN 128
#define BK 32
#define LDA_S 40   // padded LDS stride (ushorts): 32 k-values (64B) + 16B pad
#define LDB_S 40

__device__ __forceinline__ u16 f2bf(float f) {
  union { __bf16 h; u16 u; } x;
  x.h = (__bf16)f;          // native cvt, RNE
  return x.u;
}
__device__ __forceinline__ float bf2f(u16 u) {
  union { unsigned int i; float f; } x; x.i = ((unsigned int)u) << 16;
  return x.f;
}

union Frag { uint4 u[2]; v16bf v; };

// ---------------------------------------------------------------------------
// TDM: async-load a 128-row x 32-col bf16 tile (row stride = stride_elems)
// into LDS with 16B padding per 64B row (=> LDS row stride 40 ushorts).
// D# group0: count=1, lds_addr, global_addr, type=2 ("image").
// D# group1: data_size=2B, pad_enable, pad_interval=16 DW, pad_amount=4 DW,
//            tensor_dim0/1 (OOB), tile_dim0=32, tile_dim1=128, dim0 stride.
// ---------------------------------------------------------------------------
__device__ __forceinline__ void tdm_load_tile_bf16(
    unsigned int lds_byte_addr, const u16* gsrc, unsigned int stride_elems,
    unsigned int tdim0, unsigned int tdim1) {
  const unsigned long long ga = (unsigned long long)(uintptr_t)gsrc;
  v4u g0;
  g0[0] = 1u;                                                   // count = 1
  g0[1] = lds_byte_addr;                                        // lds_addr
  g0[2] = (unsigned int)ga;                                     // global_addr lo
  g0[3] = (unsigned int)((ga >> 32) & 0x1FFFFFFull) | (2u << 30); // hi | type=2
  v8i g1;
  g1[0] = (int)((1u << 16)      // data_size = 2 bytes
              | (1u << 20)      // pad_enable
              | (3u << 22)      // pad_interval: 16 DWORDs
              | (3u << 25));    // pad_amount:   4 DWORDs
  g1[1] = (int)((tdim0 & 0xFFFFu) << 16);                       // tensor_dim0 lo16
  g1[2] = (int)((tdim0 >> 16) | ((tdim1 & 0xFFFFu) << 16));     // td0 hi | td1 lo
  g1[3] = (int)((tdim1 >> 16) | (32u << 16));                   // td1 hi | tile_dim0=32
  g1[4] = (int)128u;                                            // tile_dim1=128, tile_dim2=0
  g1[5] = (int)stride_elems;                                    // tensor_dim0_stride
  g1[6] = 0;
  g1[7] = 0;
  const v4i z4 = {0, 0, 0, 0};
  const v8i z8 = {0, 0, 0, 0, 0, 0, 0, 0};
  __builtin_amdgcn_tensor_load_to_lds(g0, g1, z4, z4, z8, 0);
}

// ---------------------------------------------------------------------------
// Generic bf16 WMMA GEMM:  C = epilogue(A @ B)
//   A: M x K row-major (fp32 -> converted in staging, or bf16 -> TDM-staged)
//   B: K x N row-major bf16  (transB=1: B stored as N x K row-major, i.e. B^T)
// Block tile 128x128x32, 256 threads = 8 waves, wave tile 32x64 (2x4 WMMA).
// EPI: 0 = bf16 store, (v+bias)*alpha           (QKV projections)
//      1 = fp32 store, v*alpha, causal mask     (attention scores)
//      2 = bf16 store, gelu(v+bias)             (MLP up-proj)
//      3 = fp32 store, v+bias+residual(bf16)    (MLP down-proj + residual)
//      4 = fp32 store, v                        (attn @ V)
// ---------------------------------------------------------------------------
template<typename AT, int EPI>
__global__ __launch_bounds__(256) void gemm_wmma_kernel(
    const AT* __restrict__ Ag, const u16* __restrict__ Bg,
    const float* __restrict__ bias, const u16* __restrict__ Res,
    void* __restrict__ Cg,
    int M, int N, int K, int transB,
    long long sA, long long sB, long long sC, float alpha) {
  const int tid = threadIdx.x;
  const int bz  = blockIdx.z;
  Ag += (long long)bz * sA;
  Bg += (long long)bz * sB;
  const int lda = K;
  const int ldb = transB ? K : N;
  const int ldc = N;

  __shared__ __align__(16) u16 lA[BM * LDA_S];
  __shared__ __align__(16) u16 lB[BN * LDB_S];   // holds B^T: [n][k]

  const int w    = tid >> 5;
  const int lane = tid & 31;
  const int half = lane >> 4;
  const int lm   = lane & 15;
  const int wm   = (w & 3) * 32;   // wave row base within block tile
  const int wn   = (w >> 2) * 64;  // wave col base within block tile

  const v8f vzero = {0.f, 0.f, 0.f, 0.f, 0.f, 0.f, 0.f, 0.f};
  v8f acc[2][4];
  #pragma unroll
  for (int i = 0; i < 2; i++)
    #pragma unroll
    for (int j = 0; j < 4; j++) acc[i][j] = vzero;

  const int m0 = blockIdx.y * BM;
  const int n0 = blockIdx.x * BN;
  const bool skip = (EPI == 1) && (n0 > m0 + BM - 1);  // block fully above diagonal

  // cooperative-load thread mapping (manual staging paths)
  const int arow  = tid >> 1;         // 0..127  (A rows / B^T rows)
  const int acol  = (tid & 1) * 16;   // 0 or 16 (k offset)
  const int bkrow = tid >> 3;         // 0..31   (B k-row, transB==0)
  const int bncol = (tid & 7) * 16;   // 0..112  (B n-cols, transB==0)

  const unsigned int ldsA_base = (unsigned int)(uintptr_t)(void*)&lA[0];
  const unsigned int ldsB_base = (unsigned int)(uintptr_t)(void*)&lB[0];

  if (!skip) {
    for (int k0 = 0; k0 < K; k0 += BK) {
      // ---- stage A tile ----
      if constexpr (sizeof(AT) == 2) {
        if (w == 0)   // TDM async tile DMA (with LDS padding), wave 0 issues
          tdm_load_tile_bf16(ldsA_base, (const u16*)Ag + (long long)m0 * lda + k0,
                             (unsigned)lda, (unsigned)K, (unsigned)M);
      } else {
        __align__(16) u16 tmp[16];
        const AT* src = Ag + (long long)(m0 + arow) * lda + k0 + acol;
        #pragma unroll
        for (int i = 0; i < 4; i++) {
          float4 f = ((const float4*)src)[i];
          tmp[i*4+0] = f2bf(f.x); tmp[i*4+1] = f2bf(f.y);
          tmp[i*4+2] = f2bf(f.z); tmp[i*4+3] = f2bf(f.w);
        }
        *(uint4*)&lA[arow * LDA_S + acol]     = *(const uint4*)&tmp[0];
        *(uint4*)&lA[arow * LDA_S + acol + 8] = *(const uint4*)&tmp[8];
        if (k0 + BK < K)
          __builtin_prefetch(src + BK, 0, 1);
      }
      // ---- stage B tile transposed into LDS: lB[n][k] ----
      if (transB) {
        if constexpr (sizeof(AT) == 2) {
          if (w == 1)  // B^T rows are contiguous k-runs: same tile shape as A
            tdm_load_tile_bf16(ldsB_base, Bg + (long long)n0 * ldb + k0,
                               (unsigned)ldb, (unsigned)K, (unsigned)N);
        } else {
          const u16* src = Bg + (long long)(n0 + arow) * ldb + k0 + acol;
          *(uint4*)&lB[arow * LDB_S + acol]     = ((const uint4*)src)[0];
          *(uint4*)&lB[arow * LDB_S + acol + 8] = ((const uint4*)src)[1];
        }
      } else {
        __align__(16) u16 tb[16];
        const u16* src = Bg + (long long)(k0 + bkrow) * ldb + n0 + bncol;
        *(uint4*)&tb[0] = ((const uint4*)src)[0];
        *(uint4*)&tb[8] = ((const uint4*)src)[1];
        #pragma unroll
        for (int i = 0; i < 16; i++) lB[(bncol + i) * LDB_S + bkrow] = tb[i];
        if (k0 + BK < K)
          __builtin_prefetch(Bg + (long long)(k0 + BK + bkrow) * ldb + n0 + bncol, 0, 1);
      }
      if constexpr (sizeof(AT) == 2)
        __builtin_amdgcn_s_wait_tensorcnt(0);   // issuing waves drain TDM first
      __syncthreads();

      // ---- WMMA: 2 (M) x 4 (N) tiles of 16x16, K=32 ----
      Frag af[2];
      #pragma unroll
      for (int mi = 0; mi < 2; mi++) {
        const int m = wm + mi * 16 + lm;
        af[mi].u[0] = *(const uint4*)&lA[m * LDA_S + 8 * half];        // K = 8h..8h+7
        af[mi].u[1] = *(const uint4*)&lA[m * LDA_S + 16 + 8 * half];   // K = 16+8h..+7
      }
      #pragma unroll
      for (int ni = 0; ni < 4; ni++) {
        const int n = wn + ni * 16 + lm;
        Frag bfr;
        bfr.u[0] = *(const uint4*)&lB[n * LDB_S + 8 * half];           // K = 8h..8h+7
        bfr.u[1] = *(const uint4*)&lB[n * LDB_S + 16 + 8 * half];      // K = 16+8h..+7
        #pragma unroll
        for (int mi = 0; mi < 2; mi++) {
          acc[mi][ni] = __builtin_amdgcn_wmma_f32_16x16x32_bf16(
              false, af[mi].v, false, bfr.v, (short)0, acc[mi][ni], false, false);
        }
      }
      __syncthreads();
    }
  }

  // ---- epilogue: C/D layout -> row = r + 8*half, col = lane&15 per 16x16 tile ----
  #pragma unroll
  for (int mi = 0; mi < 2; mi++)
    #pragma unroll
    for (int ni = 0; ni < 4; ni++)
      #pragma unroll
      for (int r = 0; r < 8; r++) {
        const long long row = m0 + wm + mi * 16 + r + 8 * half;
        const long long col = n0 + wn + ni * 16 + lm;
        const float v = skip ? -1e30f : acc[mi][ni][r];
        const long long cidx = (long long)bz * sC + row * ldc + col;
        if constexpr (EPI == 0) {
          ((u16*)Cg)[cidx] = f2bf((v + bias[col]) * alpha);
        } else if constexpr (EPI == 1) {
          float o = v * alpha;
          if (col > row) o = -1e30f;
          ((float*)Cg)[cidx] = o;
        } else if constexpr (EPI == 2) {
          const float xin = v + bias[col];
          ((u16*)Cg)[cidx] = f2bf(0.5f * xin * (1.0f + erff(xin * 0.70710678118654752f)));
        } else if constexpr (EPI == 3) {
          ((float*)Cg)[cidx] = v + bias[col] + bf2f(Res[row * (long long)ldc + col]);
        } else {
          ((float*)Cg)[cidx] = v;
        }
      }
}

// ---------------------------------------------------------------------------
// Row softmax over fp32 scores (masked entries are -1e30) -> bf16 probs.
// ---------------------------------------------------------------------------
__global__ __launch_bounds__(256) void softmax_kernel(
    const float* __restrict__ S_, u16* __restrict__ P, int n) {
  const long long row = blockIdx.x;
  const float* sr = S_ + row * n;
  u16* pr = P + row * n;
  __shared__ float red[256];
  const int tid = threadIdx.x;

  float m = -1e30f;
  for (int c = tid; c < n; c += 256) m = fmaxf(m, sr[c]);
  red[tid] = m; __syncthreads();
  for (int s = 128; s > 0; s >>= 1) {
    if (tid < s) red[tid] = fmaxf(red[tid], red[tid + s]);
    __syncthreads();
  }
  m = red[0]; __syncthreads();

  float sum = 0.f;
  for (int c = tid; c < n; c += 256) sum += __expf(sr[c] - m);
  red[tid] = sum; __syncthreads();
  for (int s = 128; s > 0; s >>= 1) {
    if (tid < s) red[tid] += red[tid + s];
    __syncthreads();
  }
  const float inv = 1.f / red[0];
  for (int c = tid; c < n; c += 256) pr[c] = f2bf(__expf(sr[c] - m) * inv);
}

// ---------------------------------------------------------------------------
// Row layernorm (n=1024), OT = u16 (bf16 out) or float.
// ---------------------------------------------------------------------------
template<typename OT>
__global__ __launch_bounds__(256) void layernorm_kernel(
    const float* __restrict__ X, const float* __restrict__ g,
    const float* __restrict__ b, OT* __restrict__ Y, int n) {
  const long long row = blockIdx.x;
  const float* xr = X + row * n;
  __shared__ float s1[256], s2[256];
  const int tid = threadIdx.x;

  float a = 0.f, q = 0.f;
  for (int c = tid; c < n; c += 256) { const float v = xr[c]; a += v; q += v * v; }
  s1[tid] = a; s2[tid] = q; __syncthreads();
  for (int s = 128; s > 0; s >>= 1) {
    if (tid < s) { s1[tid] += s1[tid + s]; s2[tid] += s2[tid + s]; }
    __syncthreads();
  }
  const float mu  = s1[0] / n;
  const float var = s2[0] / n - mu * mu;
  const float inv = rsqrtf(var + 1e-5f);
  OT* yr = Y + row * n;
  for (int c = tid; c < n; c += 256) {
    const float o = (xr[c] - mu) * inv * g[c] + b[c];
    if constexpr (sizeof(OT) == 2) yr[c] = f2bf(o); else yr[c] = o;
  }
}

__global__ __launch_bounds__(256) void cvt_f32_bf16_kernel(
    const float* __restrict__ in, u16* __restrict__ out, long long n) {
  const long long i = (long long)blockIdx.x * 256 + threadIdx.x;
  if (i < n) out[i] = f2bf(in[i]);
}

// ---------------------------------------------------------------------------
extern "C" void kernel_launch(void* const* d_in, const int* in_sizes, int n_in,
                              void* d_out, int out_size, void* d_ws, size_t ws_size,
                              hipStream_t stream) {
  (void)in_sizes; (void)n_in; (void)out_size; (void)ws_size;
  const float* x   = (const float*)d_in[0];
  const float* Wq  = (const float*)d_in[1];
  const float* bq  = (const float*)d_in[2];
  const float* Wk  = (const float*)d_in[3];
  const float* bk  = (const float*)d_in[4];
  const float* Wv  = (const float*)d_in[5];
  const float* bv  = (const float*)d_in[6];
  const float* W1  = (const float*)d_in[7];
  const float* b1  = (const float*)d_in[8];
  const float* W2  = (const float*)d_in[9];
  const float* b2  = (const float*)d_in[10];
  const float* g1  = (const float*)d_in[11];
  const float* be1 = (const float*)d_in[12];
  const float* g2  = (const float*)d_in[13];
  const float* be2 = (const float*)d_in[14];

  const int Bsz = 8, S = 2048, D = 1024, DI = 4096;
  const long long MS = (long long)Bsz * S;  // 16384 rows

  char* ws = (char*)d_ws;
  size_t off = 0;
  auto alloc = [&](size_t bytes) -> void* {
    void* p = ws + off;
    off += (bytes + 255) & ~(size_t)255;
    return p;
  };
  u16*   wq_b   = (u16*)alloc((size_t)D * D * 2);
  u16*   wk_b   = (u16*)alloc((size_t)D * D * 2);
  u16*   wv_b   = (u16*)alloc((size_t)D * D * 2);
  u16*   w1_b   = (u16*)alloc((size_t)D * DI * 2);
  u16*   w2_b   = (u16*)alloc((size_t)DI * D * 2);
  u16*   q_b    = (u16*)alloc((size_t)MS * D * 2);
  u16*   k_b    = (u16*)alloc((size_t)MS * D * 2);
  u16*   v_b    = (u16*)alloc((size_t)MS * D * 2);
  float* scores = (float*)alloc((size_t)Bsz * S * S * 4);   // reused as h (same size)
  u16*   probs  = (u16*)alloc((size_t)Bsz * S * S * 2);
  float* attn   = (float*)alloc((size_t)MS * D * 4);        // reused as out_pre
  u16*   aln_b  = (u16*)alloc((size_t)MS * D * 2);
  u16*   h_b    = (u16*)scores;    // scores dead after softmax
  float* outpre = attn;            // attn dead after layernorm1

  // bf16 weights
  cvt_f32_bf16_kernel<<<dim3(D * D / 256), 256, 0, stream>>>(Wq, wq_b, (long long)D * D);
  cvt_f32_bf16_kernel<<<dim3(D * D / 256), 256, 0, stream>>>(Wk, wk_b, (long long)D * D);
  cvt_f32_bf16_kernel<<<dim3(D * D / 256), 256, 0, stream>>>(Wv, wv_b, (long long)D * D);
  cvt_f32_bf16_kernel<<<dim3(D * DI / 256), 256, 0, stream>>>(W1, w1_b, (long long)D * DI);
  cvt_f32_bf16_kernel<<<dim3(DI * D / 256), 256, 0, stream>>>(W2, w2_b, (long long)DI * D);

  const float scale = 0.03125f;  // 1/sqrt(1024), folded into Q

  // QKV projections: (16384x1024) @ (1024x1024) + bias -> bf16
  dim3 gqkv(D / BN, (unsigned)(MS / BM), 1);
  gemm_wmma_kernel<float, 0><<<gqkv, 256, 0, stream>>>(
      x, wq_b, bq, nullptr, q_b, (int)MS, D, D, 0, 0, 0, 0, scale);
  gemm_wmma_kernel<float, 0><<<gqkv, 256, 0, stream>>>(
      x, wk_b, bk, nullptr, k_b, (int)MS, D, D, 0, 0, 0, 0, 1.0f);
  gemm_wmma_kernel<float, 0><<<gqkv, 256, 0, stream>>>(
      x, wv_b, bv, nullptr, v_b, (int)MS, D, D, 0, 0, 0, 0, 1.0f);

  // scores = Q @ K^T (batched), causal mask -> fp32 (A and B^T tiles via TDM)
  dim3 gs(S / BN, S / BM, Bsz);
  gemm_wmma_kernel<u16, 1><<<gs, 256, 0, stream>>>(
      q_b, k_b, nullptr, nullptr, scores, S, S, D, 1,
      (long long)S * D, (long long)S * D, (long long)S * S, 1.0f);

  softmax_kernel<<<dim3((unsigned)MS), 256, 0, stream>>>(scores, probs, S);

  // attn = P @ V (batched) -> fp32
  dim3 ga(D / BN, S / BM, Bsz);
  gemm_wmma_kernel<u16, 4><<<ga, 256, 0, stream>>>(
      probs, v_b, nullptr, nullptr, attn, S, D, S, 0,
      (long long)S * S, (long long)S * D, (long long)S * D, 1.0f);

  layernorm_kernel<u16><<<dim3((unsigned)MS), 256, 0, stream>>>(attn, g1, be1, aln_b, D);

  // h = gelu(aln @ W1 + b1) -> bf16
  dim3 gm1(DI / BN, (unsigned)(MS / BM), 1);
  gemm_wmma_kernel<u16, 2><<<gm1, 256, 0, stream>>>(
      aln_b, w1_b, b1, nullptr, h_b, (int)MS, DI, D, 0, 0, 0, 0, 1.0f);

  // out_pre = h @ W2 + b2 + aln -> fp32
  dim3 gm2(D / BN, (unsigned)(MS / BM), 1);
  gemm_wmma_kernel<u16, 3><<<gm2, 256, 0, stream>>>(
      h_b, w2_b, b2, aln_b, outpre, (int)MS, D, DI, 0, 0, 0, 0, 1.0f);

  layernorm_kernel<float><<<dim3((unsigned)MS), 256, 0, stream>>>(
      outpre, g2, be2, (float*)d_out, D);
}
